// QGATConv_44289702756370
// MI455X (gfx1250) — matrix operand order repaired and checked by reading, not compiled, not run on hardware
//
#include <hip/hip_runtime.h>
#include <hip/hip_bf16.h>

typedef __attribute__((ext_vector_type(16))) __bf16 v16bf;
typedef __attribute__((ext_vector_type(8)))  float  v8f;

#define NEG_SLOPE 0.2f

// ---------- helpers: order-preserving float<->uint map for atomic max ----------
__device__ __forceinline__ unsigned f2ord(float f) {
    unsigned x = __float_as_uint(f);
    return (x & 0x80000000u) ? ~x : (x | 0x80000000u);
}
__device__ __forceinline__ float ord2f(unsigned x) {
    return (x & 0x80000000u) ? __uint_as_float(x & 0x7FFFFFFFu)
                             : __uint_as_float(~x);
}

// ---------- kernel 0: init output with bias, zero softmax accumulators ----------
__global__ void gat_init(float* __restrict__ out, float* __restrict__ s,
                         unsigned* __restrict__ m, const float* __restrict__ bias,
                         int N) {
    long stride = (long)gridDim.x * blockDim.x;
    long tid = (long)blockIdx.x * blockDim.x + threadIdx.x;
    long total = (long)N * 128;
    for (long i = tid; i < total; i += stride) out[i] = bias[i & 127];
    long totnh = (long)N * 4;
    for (long i = tid; i < totnh; i += stride) { s[i] = 0.0f; m[i] = 0u; }
}

// ---------- kernel 1: ft = feat @ W via bf16 WMMA, fp32 accumulate ----------
// block = 256 threads = 8 waves; each wave computes a 16x128 band of ft.
// All of W (256x128 -> bf16, 64KB) is staged once into LDS in WMMA B-fragment
// order: Bfrag[kt][t][lane][i] so a lane's fragment is 2 contiguous b128 loads.
// A fragments are loaded straight from global (one contiguous 16-float run of
// one feat row per lane, row clamped to avoid per-iteration predication) and
// converted to bf16 in registers.
__global__ __launch_bounds__(256)
void gat_gemm_bf16(const float* __restrict__ feat, const float* __restrict__ W,
                   float* __restrict__ ft, int N) {
    __shared__ __bf16 Bfrag[8][8][32][16];   // [ktile][ntile][lane][elem] = 64KB

    const int tid  = threadIdx.x;
    const int wave = tid >> 5;
    const int lane = tid & 31;
    const int r0   = blockIdx.x * 128 + wave * 16;

    // ---- stage all of W in fragment order (one pass, one barrier) ----
    // fragment-slot s -> (ktile, t, lane); each slot = 16 contiguous bf16 (32B)
    for (int s = tid; s < 2048; s += 256) {
        int ktile = s >> 8;
        int t     = (s >> 5) & 7;
        int ln    = s & 31;
        int kb    = ktile * 32 + ((ln >> 4) << 4);  // K base for this lane half
        int c     = t * 16 + (ln & 15);             // output column
        union { __bf16 b[16]; uint4 u[2]; } tmp;
        #pragma unroll
        for (int i = 0; i < 16; ++i)
            tmp.b[i] = (__bf16)W[(kb + i) * 128 + c];
        uint4* dp = (uint4*)&Bfrag[ktile][t][ln][0];
        dp[0] = tmp.u[0];
        dp[1] = tmp.u[1];
    }
    __syncthreads();

    v8f acc[8] = {};

    // A-fragment geometry (contiguous k order used here; Bfrag staged to match):
    // lanes 0-15: M=lane,    K in {kgrp..kgrp+15}, kgrp=0
    // lanes 16-31: M=lane-16, same pattern with kgrp=16
    const int am   = lane & 15;
    const int kgrp = (lane >> 4) * 16;
    int row = r0 + am;
    if (row > N - 1) row = N - 1;          // clamp: OOB lanes compute a valid
    const float* fr = feat + (long)row * 256;  // (unstored) row -> no branching

    for (int kt = 0; kt < 8; ++kt) {
        const int base = kt * 32 + kgrp;
        float fa[16] __attribute__((aligned(16)));
        *(float4*)&fa[0]  = *(const float4*)(fr + base);
        *(float4*)&fa[4]  = *(const float4*)(fr + base + 4);
        *(float4*)&fa[8]  = *(const float4*)(fr + base + 8);
        *(float4*)&fa[12] = *(const float4*)(fr + base + 12);

        union { __bf16 b[16]; v16bf v; } af;
        #pragma unroll
        for (int i = 0; i < 16; ++i) af.b[i] = (__bf16)fa[i];

        #pragma unroll
        for (int t = 0; t < 8; ++t) {
            union { uint4 u[2]; v16bf v; } bf;
            const uint4* bp = (const uint4*)&Bfrag[kt][t][lane][0];
            bf.u[0] = bp[0];
            bf.u[1] = bp[1];
            acc[t] = __builtin_amdgcn_wmma_f32_16x16x32_bf16(
                false, af.v, false, bf.v, (short)0, acc[t], false, false);
        }
    }

    // C/D layout: VGPR v -> M = v + 8*(lane/16); N = lane&15
    const int ncol  = lane & 15;
    const int rbase = (lane >> 4) * 8;
    #pragma unroll
    for (int t = 0; t < 8; ++t) {
        #pragma unroll
        for (int v = 0; v < 8; ++v) {
            int orow = r0 + rbase + v;
            if (orow < N) ft[(long)orow * 128 + t * 16 + ncol] = acc[t][v];
        }
    }
}

// ---------- kernel 2: per-node attention logits el/er ----------
__global__ void gat_dots(const float* __restrict__ ft,
                         const float* __restrict__ al, const float* __restrict__ ar,
                         float* __restrict__ el, float* __restrict__ er, int N) {
    int idx = blockIdx.x * blockDim.x + threadIdx.x;  // n*4 + h
    if (idx >= N * 4) return;
    int n = idx >> 2, h = idx & 3;
    const float* f = ft + (long)n * 128 + h * 32;
    float sl = 0.0f, sr = 0.0f;
    #pragma unroll
    for (int d = 0; d < 32; ++d) {
        float v = f[d];
        sl += v * al[h * 32 + d];
        sr += v * ar[h * 32 + d];
    }
    el[idx] = sl;
    er[idx] = sr;
}

// ---------- kernel 3: edge scores + segment max ----------
__global__ void gat_edge_score(const int* __restrict__ src, const int* __restrict__ dst,
                               const float* __restrict__ el, const float* __restrict__ er,
                               float* __restrict__ esc, unsigned* __restrict__ m, int E) {
    int idx = blockIdx.x * blockDim.x + threadIdx.x;  // e*4 + h
    if (idx >= E * 4) return;
    int e = idx >> 2, h = idx & 3;
    float v = el[src[e] * 4 + h] + er[dst[e] * 4 + h];
    v = (v > 0.0f) ? v : NEG_SLOPE * v;               // leaky relu
    esc[idx] = v;
    atomicMax(&m[dst[e] * 4 + h], f2ord(v));
}

// ---------- kernel 4: exp(e - max) + segment sum ----------
__global__ void gat_edge_expsum(const int* __restrict__ dst, float* __restrict__ esc,
                                const unsigned* __restrict__ m, float* __restrict__ s, int E) {
    int idx = blockIdx.x * blockDim.x + threadIdx.x;
    if (idx >= E * 4) return;
    int e = idx >> 2, h = idx & 3;
    float w = __expf(esc[idx] - ord2f(m[dst[e] * 4 + h]));
    esc[idx] = w;
    atomicAdd(&s[dst[e] * 4 + h], w);
}

// ---------- kernel 5: weighted aggregation (SpMM u_mul_e) ----------
// one wave per edge; lane covers one column per head
__global__ __launch_bounds__(256)
void gat_aggregate(const int* __restrict__ src, const int* __restrict__ dst,
                   const float* __restrict__ esc, const float* __restrict__ s,
                   const float* __restrict__ ft, float* __restrict__ out, int E) {
    int tid = blockIdx.x * blockDim.x + threadIdx.x;
    int e = tid >> 5;
    int lane = tid & 31;
    if (e >= E) return;
    int u = src[e], v = dst[e];
    #pragma unroll
    for (int h = 0; h < 4; ++h) {
        float a = esc[e * 4 + h] / s[v * 4 + h];
        int c = h * 32 + lane;
        atomicAdd(&out[(long)v * 128 + c], ft[(long)u * 128 + c] * a);
    }
}

extern "C" void kernel_launch(void* const* d_in, const int* in_sizes, int n_in,
                              void* d_out, int out_size, void* d_ws, size_t ws_size,
                              hipStream_t stream) {
    const float* feat = (const float*)d_in[0];
    const int*   src  = (const int*)d_in[1];
    const int*   dst  = (const int*)d_in[2];
    const float* W    = (const float*)d_in[3];
    const float* al   = (const float*)d_in[4];
    const float* ar   = (const float*)d_in[5];
    const float* bias = (const float*)d_in[6];

    const int N = in_sizes[0] / 256;   // 50000
    const int E = in_sizes[1];         // 800000
    float* out = (float*)d_out;

    // workspace layout
    char* p = (char*)d_ws;
    float*    ft  = (float*)p;    p += (size_t)N * 128 * sizeof(float);
    float*    el  = (float*)p;    p += (size_t)N * 4 * sizeof(float);
    float*    er  = (float*)p;    p += (size_t)N * 4 * sizeof(float);
    unsigned* m   = (unsigned*)p; p += (size_t)N * 4 * sizeof(unsigned);
    float*    s   = (float*)p;    p += (size_t)N * 4 * sizeof(float);
    float*    esc = (float*)p;    // E*4 floats

    gat_init<<<1024, 256, 0, stream>>>(out, s, m, bias, N);
    gat_gemm_bf16<<<(N + 127) / 128, 256, 0, stream>>>(feat, W, ft, N);
    gat_dots<<<(N * 4 + 255) / 256, 256, 0, stream>>>(ft, al, ar, el, er, N);
    gat_edge_score<<<(E * 4 + 255) / 256, 256, 0, stream>>>(src, dst, el, er, esc, m, E);
    gat_edge_expsum<<<(E * 4 + 255) / 256, 256, 0, stream>>>(dst, esc, m, s, E);
    gat_aggregate<<<(E + 7) / 8, 256, 0, stream>>>(src, dst, esc, s, ft, out, E);
}
